// SimpleMultiHeadTemporalAttention_57947698758272
// MI455X (gfx1250) — compile-verified, matrix operand order checked by reading, not run on hardware
//
#include <hip/hip_runtime.h>
#include <hip/hip_bf16.h>
#include <hip/hip_fp16.h>

// Shapes from the reference
#define BDIM 8
#define SDIM 1024
#define DDIM 1024
#define HDIM 16
#define HD   64

typedef __attribute__((ext_vector_type(16))) _Float16 v16h;
typedef __attribute__((ext_vector_type(8)))  _Float16 h8;
typedef __attribute__((ext_vector_type(8)))  float    v8f;
typedef __attribute__((ext_vector_type(4)))  float    f4;

union V16H { v16h v; h8 h[2]; _Float16 e[16]; };

__device__ __forceinline__ v8f wmma16(v16h a, v16h b, v8f c) {
  // D = A(16x32 f16) * B(32x16 f16) + C(16x16 f32)
  return __builtin_amdgcn_wmma_f32_16x16x32_f16(
      /*neg_a=*/false, a, /*neg_b=*/false, b,
      /*c_mod=*/(short)0, c, /*reuse_a=*/false, /*reuse_b=*/false);
}

// ---------------------------------------------------------------- stage 0
__global__ void cvt_f32_to_f16(const float* __restrict__ src,
                               _Float16* __restrict__ dst, int n) {
  int i = blockIdx.x * blockDim.x + threadIdx.x;
  if (i < n) dst[i] = (_Float16)src[i];
}

// Convert + per-head transpose: src (h, d, e) f32 -> dst (h, e, d) f16,
// so a WMMA B-operand column (fixed e) is contiguous in k.
__global__ void cvt_w_qkv_t(const float* __restrict__ src,
                            _Float16* __restrict__ dst, int n) {
  int i = blockIdx.x * blockDim.x + threadIdx.x;
  if (i >= n) return;
  const int h   = i >> 12;      // / (64*64)
  const int rem = i & 4095;
  const int d   = rem >> 6;     // input dim
  const int e   = rem & 63;     // output dim
  dst[(h << 12) + (e << 6) + d] = (_Float16)src[i];
}

// ---------------------------------------------------------------- stage 1
// Per-head QKV projection. Block = 4 waves; wave w -> output cols [16w,16w+16).
// q pre-scaled by 1/sqrt(HD); v stored TRANSPOSED (b,h,e,s) for stage 2.
__global__ void __launch_bounds__(128)
qkv_proj_kernel(const float* __restrict__ x,
                const _Float16* __restrict__ wqT,
                const _Float16* __restrict__ wkT,
                const _Float16* __restrict__ wvT,
                const float* __restrict__ bq,
                const float* __restrict__ bk,
                const float* __restrict__ bv,
                _Float16* __restrict__ q16,
                _Float16* __restrict__ k16,
                _Float16* __restrict__ vT16) {
  const int s0   = blockIdx.x * 16;
  const int h    = blockIdx.y;
  const int b    = blockIdx.z;
  const int wave = threadIdx.x >> 5;
  const int lane = threadIdx.x & 31;
  const int half = lane >> 4;
  const int lr   = lane & 15;

  // A tile: xh rows (16 x 64), vector f32 loads + pack to f16
  V16H a[2];
  const float* xrow = x + ((size_t)b * SDIM + (s0 + lr)) * DDIM + h * HD;
#pragma unroll
  for (int kb = 0; kb < 2; ++kb) {
    const float* c0 = xrow + kb * 32 + half * 8;        // K slots j=0..7
    const float* c1 = xrow + kb * 32 + 16 + half * 8;   // K slots j=8..15
    f4 u0 = *(const f4*)(c0);
    f4 u1 = *(const f4*)(c0 + 4);
    f4 u2 = *(const f4*)(c1);
    f4 u3 = *(const f4*)(c1 + 4);
#pragma unroll
    for (int j = 0; j < 4; ++j) {
      a[kb].e[j]      = (_Float16)u0[j];
      a[kb].e[4 + j]  = (_Float16)u1[j];
      a[kb].e[8 + j]  = (_Float16)u2[j];
      a[kb].e[12 + j] = (_Float16)u3[j];
    }
  }

  const int e = wave * 16 + lr;  // output feature column
  const _Float16* Ws[3] = {wqT, wkT, wvT};
  const float*    Bs[3] = {bq, bk, bv};
  _Float16*       Os[3] = {q16, k16, vT16};
  const size_t obase = ((size_t)b * HDIM + h) * SDIM * HD;

#pragma unroll
  for (int m = 0; m < 3; ++m) {
    // B column e = contiguous row e of W^T
    const _Float16* wrow = Ws[m] + ((size_t)h * HD + e) * HD;
    v8f acc = {};
#pragma unroll
    for (int kb = 0; kb < 2; ++kb) {
      V16H bop;
      bop.h[0] = *(const h8*)(wrow + kb * 32 + half * 8);
      bop.h[1] = *(const h8*)(wrow + kb * 32 + 16 + half * 8);
      acc = wmma16(a[kb].v, bop.v, acc);
    }
    const float bias = Bs[m][h * HD + e];
    const float scl  = (m == 0) ? 0.125f : 1.0f;  // 1/sqrt(64) folded into q
#pragma unroll
    for (int r = 0; r < 8; ++r) {
      const int row = s0 + r + 8 * half;
      const _Float16 val = (_Float16)((acc[r] + bias) * scl);
      if (m == 2)
        Os[m][obase + (size_t)e * SDIM + row] = val;      // v transposed (e,s)
      else
        Os[m][obase + (size_t)row * HD + e] = val;        // q,k as (s,e)
    }
  }
}

// ---------------------------------------------------------------- stage 2
// Flash attention, one wave per (b, h, 16-query tile); 64 keys per step.
// lrun accumulated LANE-LOCALLY (uniform alpha) -> one cross-lane reduce at end.
__global__ void __launch_bounds__(32)
attn_kernel(const _Float16* __restrict__ q16,
            const _Float16* __restrict__ k16,
            const _Float16* __restrict__ vT16,
            const unsigned char* __restrict__ mask,  // True = padded key
            _Float16* __restrict__ concat) {
  __shared__ _Float16 Plds[16 * 64];  // P tile, row-major 16 rows x 64 keys

  const int s0   = blockIdx.x * 16;
  const int h    = blockIdx.y;
  const int b    = blockIdx.z;
  const int lane = threadIdx.x & 31;
  const int half = lane >> 4;
  const int lr   = lane & 15;

  const size_t bh = (size_t)b * HDIM + h;
  const _Float16* qb  = q16 + bh * SDIM * HD;
  const _Float16* kbp = k16 + bh * SDIM * HD;
  const _Float16* vbT = vT16 + bh * SDIM * HD;  // (e, s) layout
  const unsigned char* maskrow = mask + (size_t)b * SDIM;

  // q A-tile: resident for the whole kernel
  V16H aq[2];
  const _Float16* qrow = qb + (size_t)(s0 + lr) * HD;
#pragma unroll
  for (int kb = 0; kb < 2; ++kb) {
    aq[kb].h[0] = *(const h8*)(qrow + kb * 32 + half * 8);
    aq[kb].h[1] = *(const h8*)(qrow + kb * 32 + 16 + half * 8);
  }

  // per-e-tile V^T row pointers (column e = et*16+lr of V)
  const _Float16* vcol[4];
#pragma unroll
  for (int et = 0; et < 4; ++et)
    vcol[et] = vbT + (size_t)(et * 16 + lr) * SDIM;

  float mrun[8], lrun[8];
  v8f acc[4];
#pragma unroll
  for (int r = 0; r < 8; ++r) { mrun[r] = -INFINITY; lrun[r] = 0.0f; }
#pragma unroll
  for (int t = 0; t < 4; ++t) acc[t] = (v8f){};

  for (int kt0 = 0; kt0 < SDIM; kt0 += 64) {
    // ---- scores: four 16-key tiles (q already carries the 1/sqrt(HD) scale)
    v8f sc[4];
#pragma unroll
    for (int t = 0; t < 4; ++t) {
      const int key = kt0 + t * 16 + lr;  // this lane's key column
      const _Float16* krow = kbp + (size_t)key * HD;
      v8f s = {};
#pragma unroll
      for (int kb = 0; kb < 2; ++kb) {
        V16H bkt;  // B = K^T, column = key row (contiguous 16B chunks)
        bkt.h[0] = *(const h8*)(krow + kb * 32 + half * 8);
        bkt.h[1] = *(const h8*)(krow + kb * 32 + 16 + half * 8);
        s = wmma16(aq[kb].v, bkt.v, s);
      }
      // key-padding mask -> finfo(f32).min (matches reference semantics,
      // including the all-masked-row -> uniform-softmax corner case)
      if (maskrow[key]) {
#pragma unroll
        for (int r = 0; r < 8; ++r) s[r] = -3.4028235e38f;
      }
      sc[t] = s;
    }

    // ---- prefetch next 64-key K/V chunk into caches (global_prefetch_b8)
    if (kt0 + 64 < SDIM) {
      __builtin_prefetch(kbp + (size_t)(kt0 + 64 + lane) * HD, 0, 3);
      __builtin_prefetch(kbp + (size_t)(kt0 + 96 + lane) * HD, 0, 3);
      __builtin_prefetch(vbT + (size_t)lane * SDIM + kt0 + 64, 0, 3);
      __builtin_prefetch(vbT + (size_t)(32 + lane) * SDIM + kt0 + 64, 0, 3);
    }

    // ---- online softmax: row max across tiles, then across 16 lanes
    float mnew[8];
#pragma unroll
    for (int r = 0; r < 8; ++r) {
      float mx = fmaxf(fmaxf(sc[0][r], sc[1][r]), fmaxf(sc[2][r], sc[3][r]));
#pragma unroll
      for (int d = 1; d < 16; d <<= 1) mx = fmaxf(mx, __shfl_xor(mx, d, 32));
      mnew[r] = fmaxf(mrun[r], mx);
    }
    float alpha[8];
#pragma unroll
    for (int r = 0; r < 8; ++r) alpha[r] = __expf(mrun[r] - mnew[r]);
#pragma unroll
    for (int t = 0; t < 4; ++t)
#pragma unroll
      for (int r = 0; r < 8; ++r) acc[t][r] *= alpha[r];
#pragma unroll
    for (int r = 0; r < 8; ++r) {
      float psum = 0.0f;
#pragma unroll
      for (int t = 0; t < 4; ++t) {
        const float p = __expf(sc[t][r] - mnew[r]);
        Plds[(r + 8 * half) * 64 + t * 16 + lr] = (_Float16)p;
        psum += p;
      }
      lrun[r] = lrun[r] * alpha[r] + psum;  // lane-local partial sum
      mrun[r] = mnew[r];
    }
    __syncthreads();  // C-layout -> A-layout relayout through LDS

    V16H pa[2];
    const _Float16* prow = Plds + lr * 64;
#pragma unroll
    for (int kc = 0; kc < 2; ++kc) {
      pa[kc].h[0] = *(const h8*)(prow + kc * 32 + half * 8);
      pa[kc].h[1] = *(const h8*)(prow + kc * 32 + 16 + half * 8);
    }
    __syncthreads();

    // ---- acc += P(16x64) x V(64x64): B from transposed V, 16B loads
#pragma unroll
    for (int et = 0; et < 4; ++et) {
#pragma unroll
      for (int kc = 0; kc < 2; ++kc) {
        V16H bvv;
        bvv.h[0] = *(const h8*)(vcol[et] + kt0 + kc * 32 + half * 8);
        bvv.h[1] = *(const h8*)(vcol[et] + kt0 + kc * 32 + 16 + half * 8);
        acc[et] = wmma16(pa[kc].v, bvv.v, acc[et]);
      }
    }
  }

  // ---- final denominator: single cross-lane reduction, then store f16
#pragma unroll
  for (int r = 0; r < 8; ++r) {
    float l = lrun[r];
#pragma unroll
    for (int d = 1; d < 16; d <<= 1) l += __shfl_xor(l, d, 32);
    const float inv = 1.0f / l;
    const int row = s0 + r + 8 * half;
#pragma unroll
    for (int et = 0; et < 4; ++et) {
      const size_t o =
          ((size_t)b * SDIM + row) * DDIM + h * HD + et * 16 + lr;
      concat[o] = (_Float16)(acc[et][r] * inv);
    }
  }
}

// ---------------------------------------------------------------- stage 3
// out = concat(8192x1024) @ Wo^T + bo.
// Block = 4 waves; each wave computes a 32x32 output tile (4 WMMAs / 8 loads).
__global__ void __launch_bounds__(128)
out_proj_kernel(const _Float16* __restrict__ concat,
                const _Float16* __restrict__ wo16,  // (n, k) row-major
                const float* __restrict__ bo,
                float* __restrict__ out) {
  const int m0   = blockIdx.y * 32;
  const int wave = threadIdx.x >> 5;
  const int n0   = (blockIdx.x * 4 + wave) * 32;
  const int lane = threadIdx.x & 31;
  const int half = lane >> 4;
  const int lr   = lane & 15;

  const _Float16* arow0 = concat + (size_t)(m0 + lr) * DDIM;
  const _Float16* arow1 = arow0 + (size_t)16 * DDIM;
  const _Float16* brow0 = wo16 + (size_t)(n0 + lr) * DDIM;  // B col n = Wo row n
  const _Float16* brow1 = brow0 + (size_t)16 * DDIM;
  v8f acc00 = {}, acc01 = {}, acc10 = {}, acc11 = {};
  for (int k = 0; k < DDIM; k += 32) {
    V16H a0, a1, b0, b1;
    a0.h[0] = *(const h8*)(arow0 + k + half * 8);
    a0.h[1] = *(const h8*)(arow0 + k + 16 + half * 8);
    a1.h[0] = *(const h8*)(arow1 + k + half * 8);
    a1.h[1] = *(const h8*)(arow1 + k + 16 + half * 8);
    b0.h[0] = *(const h8*)(brow0 + k + half * 8);
    b0.h[1] = *(const h8*)(brow0 + k + 16 + half * 8);
    b1.h[0] = *(const h8*)(brow1 + k + half * 8);
    b1.h[1] = *(const h8*)(brow1 + k + 16 + half * 8);
    acc00 = wmma16(a0.v, b0.v, acc00);
    acc01 = wmma16(a0.v, b1.v, acc01);
    acc10 = wmma16(a1.v, b0.v, acc10);
    acc11 = wmma16(a1.v, b1.v, acc11);
  }
  const float bias0 = bo[n0 + lr];
  const float bias1 = bo[n0 + 16 + lr];
#pragma unroll
  for (int r = 0; r < 8; ++r) {
    const int mA = m0 + r + 8 * half;
    const int mB = mA + 16;
    out[(size_t)mA * DDIM + n0 + lr]      = acc00[r] + bias0;
    out[(size_t)mA * DDIM + n0 + 16 + lr] = acc01[r] + bias1;
    out[(size_t)mB * DDIM + n0 + lr]      = acc10[r] + bias0;
    out[(size_t)mB * DDIM + n0 + 16 + lr] = acc11[r] + bias1;
  }
}

// ---------------------------------------------------------------- launch
extern "C" void kernel_launch(void* const* d_in, const int* in_sizes, int n_in,
                              void* d_out, int out_size, void* d_ws,
                              size_t ws_size, hipStream_t stream) {
  const float* x            = (const float*)d_in[0];
  const unsigned char* mask = (const unsigned char*)d_in[1];  // bool, 1 byte
  const float* Wq = (const float*)d_in[2];
  const float* bq = (const float*)d_in[3];
  const float* Wk = (const float*)d_in[4];
  const float* bk = (const float*)d_in[5];
  const float* Wv = (const float*)d_in[6];
  const float* bv = (const float*)d_in[7];
  const float* Wo = (const float*)d_in[8];
  const float* bo = (const float*)d_in[9];
  float* out = (float*)d_out;

  // f16 workspace carve-out (all offsets 16B-aligned)
  _Float16* ws = (_Float16*)d_ws;
  size_t off = 0;
  _Float16* q16  = ws + off; off += (size_t)BDIM * HDIM * SDIM * HD;
  _Float16* k16  = ws + off; off += (size_t)BDIM * HDIM * SDIM * HD;
  _Float16* vT16 = ws + off; off += (size_t)BDIM * HDIM * SDIM * HD;
  _Float16* cc   = ws + off; off += (size_t)BDIM * SDIM * DDIM;
  _Float16* wqT  = ws + off; off += (size_t)HDIM * HD * HD;
  _Float16* wkT  = ws + off; off += (size_t)HDIM * HD * HD;
  _Float16* wvT  = ws + off; off += (size_t)HDIM * HD * HD;
  _Float16* wo16 = ws + off; off += (size_t)DDIM * DDIM;

  const int nW  = HDIM * HD * HD;
  const int nWo = DDIM * DDIM;
  cvt_w_qkv_t<<<(nW + 255) / 256, 256, 0, stream>>>(Wq, wqT, nW);
  cvt_w_qkv_t<<<(nW + 255) / 256, 256, 0, stream>>>(Wk, wkT, nW);
  cvt_w_qkv_t<<<(nW + 255) / 256, 256, 0, stream>>>(Wv, wvT, nW);
  cvt_f32_to_f16<<<(nWo + 255) / 256, 256, 0, stream>>>(Wo, wo16, nWo);

  qkv_proj_kernel<<<dim3(SDIM / 16, HDIM, BDIM), 128, 0, stream>>>(
      x, wqT, wkT, wvT, bq, bk, bv, q16, k16, vT16);

  attn_kernel<<<dim3(SDIM / 16, HDIM, BDIM), 32, 0, stream>>>(
      q16, k16, vT16, mask, cc);

  out_proj_kernel<<<dim3(DDIM / 128, (BDIM * SDIM) / 32), 128, 0, stream>>>(
      cc, wo16, bo, out);
}
